// Autotuner_FFN_2078764171803
// MI455X (gfx1250) — matrix operand order repaired and checked by reading, not compiled, not run on hardware
//
#include <hip/hip_runtime.h>
#include <hip/hip_bf16.h>
#include <stdint.h>

// ---------------- types ----------------
typedef __bf16 bf16_t;
typedef __attribute__((ext_vector_type(16))) __bf16 v16bf;
typedef __attribute__((ext_vector_type(8)))  __bf16 v8bf;
typedef __attribute__((ext_vector_type(8)))  float  v8f;

union Frag { v16bf v; v8bf h[2]; };

// ---------------- constants ----------------
#define FEATN    1115
#define FPAD     1120      // padded K for layer 1 (multiple of 32)
#define FSTR     1128      // LDS row stride (bf16 elems) for feature tile
#define HN       1024
#define HSTR     1032      // LDS row stride (bf16 elems) for hidden tile
#define ROWS     64        // rows per block
#define RT       4         // 16-row tiles per block
#define CT       4         // 16-col tiles per wave (64-col strip)
#define THREADS  512       // 16 wave32s
#define EPSV     1e-5f

__device__ __forceinline__ bf16_t f2bf(float x) {
    union { float f; uint32_t u; } a; a.f = x;
    uint32_t r = a.u + 0x7FFFu + ((a.u >> 16) & 1u);   // round-to-nearest-even
    union { bf16_t b; uint16_t s; } c; c.s = (uint16_t)(r >> 16);
    return c.b;
}
__device__ __forceinline__ float bf2f(bf16_t b) {
    union { bf16_t b; uint16_t s; } c; c.b = b;
    union { float f; uint32_t u; } a; a.u = ((uint32_t)c.s) << 16;
    return a.f;
}

// ds_swizzle_b32 XOR-mode lane swap: imm = (xor_mask<<10) | (or=0)<<5 | and_mask=0x1f
template<int XORM>
__device__ __forceinline__ float swz_xor(float x) {
    union { float f; int i; } u; u.f = x;
    u.i = __builtin_amdgcn_ds_swizzle(u.i, (XORM << 10) | 0x1f);
    return u.f;
}

// ---------------- weight transpose+convert pre-pass ----------------
__global__ __launch_bounds__(256) void convert_w1(const float* __restrict__ W1,
                                                  bf16_t* __restrict__ Wt1) {
    int idx = blockIdx.x * 256 + threadIdx.x;          // over 1024*1120
    if (idx >= HN * FPAD) return;
    int n = idx / FPAD, k = idx - n * FPAD;
    float v = (k < FEATN) ? W1[(size_t)k * HN + n] : 0.0f;
    Wt1[idx] = f2bf(v);
}
__global__ __launch_bounds__(256) void convert_w2(const float* __restrict__ W2,
                                                  bf16_t* __restrict__ Wt2) {
    int idx = blockIdx.x * 256 + threadIdx.x;          // over 1024*1024
    if (idx >= HN * HN) return;
    int n = idx >> 10, k = idx & 1023;
    Wt2[idx] = f2bf(W2[(size_t)k * HN + n]);
}

// ---------------- one fused MLP layer: GEMM + bias + LN + ReLU ----------------
// A: [64 x K] bf16 in LDS (row stride aStride). Wt: [1024 x K] bf16 (global, K-major).
// Out: [64 x 1024] bf16 in LDS (row stride oStride), after layernorm+relu.
// Wave w: all 4 row tiles x 64-col strip [w*64, w*64+64).
__device__ __forceinline__ void mlp_layer(
    const bf16_t* __restrict__ sA, int aStride, int K,
    const bf16_t* __restrict__ Wt, int kStride,
    const float* __restrict__ bias, const float* __restrict__ gamma,
    const float* __restrict__ beta,
    bf16_t* __restrict__ sOut, int oStride,
    float* sSum, float* sSq, int lane, int wave)
{
    const int colBase = wave * (CT * 16);     // 64-col strip
    const int r  = lane & 15;
    const int hh = lane >> 4;

    const bf16_t* aRow[RT];
    #pragma unroll
    for (int rt = 0; rt < RT; ++rt)
        aRow[rt] = sA + (size_t)(rt * 16 + r) * aStride;

    const bf16_t* bRow[CT];
    #pragma unroll
    for (int ct = 0; ct < CT; ++ct)
        bRow[ct] = Wt + (size_t)(colBase + ct * 16 + r) * kStride + hh * 16;

    v8f acc[CT][RT];
    #pragma unroll
    for (int ct = 0; ct < CT; ++ct)
        #pragma unroll
        for (int rt = 0; rt < RT; ++rt)
            #pragma unroll
            for (int i = 0; i < 8; ++i) acc[ct][rt][i] = 0.0f;

    #pragma unroll 1
    for (int k0 = 0; k0 < K; k0 += 32) {
        // A fragments: 16x32 bf16, contiguous K [h*16, h*16+16) to match B
        Frag a[RT];
        #pragma unroll
        for (int rt = 0; rt < RT; ++rt) {
            a[rt].h[0] = *(const v8bf*)(aRow[rt] + k0 + hh * 16);
            a[rt].h[1] = *(const v8bf*)(aRow[rt] + k0 + hh * 16 + 8);
        }
        #pragma unroll
        for (int ct = 0; ct < CT; ++ct) {
            // B fragment: column n = colBase+ct*16+r, contiguous K [h*16, h*16+16)
            const bf16_t* bp = bRow[ct] + k0;
            Frag b;
            b.h[0] = *(const v8bf*)(bp);
            b.h[1] = *(const v8bf*)(bp + 8);
            #pragma unroll
            for (int rt = 0; rt < RT; ++rt)
                acc[ct][rt] = __builtin_amdgcn_wmma_f32_16x16x32_bf16(
                    false, a[rt].v, false, b.v, (short)0, acc[ct][rt], false, false);
        }
    }

    // bias add + per-row sum / sumsq.
    // For fixed (rt,v) all CT accumulator entries belong to the same row m:
    // pre-reduce over ct in registers, then one 4-step ds_swizzle XOR reduction
    // over the 16-lane group, then a single LDS atomic per (rt,v).
    float bs[CT];
    #pragma unroll
    for (int ct = 0; ct < CT; ++ct) bs[ct] = bias[colBase + ct * 16 + r];

    #pragma unroll
    for (int rt = 0; rt < RT; ++rt) {
        #pragma unroll
        for (int v = 0; v < 8; ++v) {
            float s = 0.0f, q = 0.0f;
            #pragma unroll
            for (int ct = 0; ct < CT; ++ct) {
                float x = acc[ct][rt][v] + bs[ct];
                acc[ct][rt][v] = x;
                s += x;
                q = fmaf(x, x, q);
            }
            s += swz_xor<1>(s);  q += swz_xor<1>(q);
            s += swz_xor<2>(s);  q += swz_xor<2>(q);
            s += swz_xor<4>(s);  q += swz_xor<4>(q);
            s += swz_xor<8>(s);  q += swz_xor<8>(q);
            if (r == 0) {
                int m = rt * 16 + hh * 8 + v;
                atomicAdd(&sSum[m], s);
                atomicAdd(&sSq[m],  q);
            }
        }
    }
    __syncthreads();

    // finalize stats: sSum[m] <- mu, sSq[m] <- rsqrt(var+eps)
    {
        int t = wave * 32 + lane;
        if (t < ROWS) {
            float mu  = sSum[t] * (1.0f / (float)HN);
            float var = sSq[t] * (1.0f / (float)HN) - mu * mu;
            sSum[t] = mu;
            sSq[t]  = rsqrtf(var + EPSV);
        }
    }
    __syncthreads();

    // apply layernorm + relu, write bf16 tile
    #pragma unroll
    for (int ct = 0; ct < CT; ++ct) {
        int n = colBase + ct * 16 + r;
        float gm = gamma[n], bt = beta[n];
        #pragma unroll
        for (int rt = 0; rt < RT; ++rt) {
            #pragma unroll
            for (int v = 0; v < 8; ++v) {
                int m = rt * 16 + hh * 8 + v;
                float x = (acc[ct][rt][v] - sSum[m]) * sSq[m] * gm + bt;
                x = fmaxf(x, 0.0f);
                sOut[(size_t)m * oStride + n] = f2bf(x);
            }
        }
    }
    __syncthreads();
}

// ---------------- fused kernel ----------------
extern "C" __global__ __launch_bounds__(THREADS)
void fused_mlp(const int* __restrict__ kc, const int* __restrict__ nl,
               const float* __restrict__ op_vec, const float* __restrict__ shints,
               const float* __restrict__ rdf, const int* __restrict__ rdb,
               const float* __restrict__ wdf, const int* __restrict__ wdb,
               const float* __restrict__ rest,
               const float* __restrict__ Ekc, const float* __restrict__ Enl,
               const float* __restrict__ opW, const float* __restrict__ opB,
               const float* __restrict__ Ec, const float* __restrict__ Es,
               const float* __restrict__ Ei,
               const float* __restrict__ b1, const float* __restrict__ b2,
               const float* __restrict__ W3, const float* __restrict__ b3,
               const float* __restrict__ g1, const float* __restrict__ be1,
               const float* __restrict__ g2, const float* __restrict__ be2,
               const bf16_t* __restrict__ Wt1, const bf16_t* __restrict__ Wt2,
               float* __restrict__ out, int nrows)
{
    extern __shared__ char smem[];
    bf16_t* sFeat = (bf16_t*)smem;                                   // 64 x 1128
    bf16_t* sH    = (bf16_t*)(smem + (size_t)ROWS * FSTR * 2);       // 64 x 1032
    float*  sSum  = (float*)(smem + (size_t)ROWS * FSTR * 2 + (size_t)ROWS * HSTR * 2);
    float*  sSq   = sSum + ROWS;
    float*  sRed  = sSq + ROWS;                                      // THREADS floats

    const int tid  = threadIdx.x;
    const int lane = tid & 31, wave = tid >> 5;
    const int row0 = blockIdx.x * ROWS;

    if (tid < 2 * ROWS) sSum[tid] = 0.0f;   // zero sSum+sSq

    // ---- build 64 x 1120 bf16 feature tile (8 threads per row, no div/mod) ----
    {
        const int row = tid >> 3;         // 0..63
        const int p   = tid & 7;
        const int g   = row0 + row;
        const bool ok = (g < nrows);
        bf16_t* frow = sFeat + (size_t)row * FSTR;
        for (int f = p; f < FPAD; f += 8) {
            float val = 0.0f;
            if (ok) {
                if (f < 16) {
                    val = Ekc[kc[g] * 16 + f];
                } else if (f < 32) {
                    val = Enl[nl[g] * 16 + (f - 16)];
                } else if (f < 944) {
                    int q = f - 32; int i = q >> 4; int j = q & 15;
                    val = op_vec[(size_t)g * 57 + i] * opW[i * 16 + j] + opB[i * 16 + j];
                } else if (f < 947) {
                    val = log2f(shints[(size_t)g * 3 + (f - 944)] + 1.0f);
                } else if (f < 1107) {
                    int q = f - 947;
                    const float* df = rdf; const int* db = rdb;
                    if (q >= 80) { df = wdf; db = wdb; q -= 80; }
                    int d = q / 20, o = q - d * 20;
                    size_t base = ((size_t)g * 4 + d);
                    if (o < 2) {
                        val = df[base * 8 + o];
                    } else if (o < 8) {
                        float ss = df[base * 8 + o];
                        val = copysignf(log2f(fabsf(ss) + 1.0f), ss);
                    } else if (o < 12) {
                        val = Ec[db[base * 3 + 0] * 4 + (o - 8)];
                    } else if (o < 16) {
                        val = Es[db[base * 3 + 1] * 4 + (o - 12)];
                    } else {
                        val = Ei[db[base * 3 + 2] * 4 + (o - 16)];
                    }
                } else if (f < 1115) {
                    int q = f - 1107;
                    float rv = rest[(size_t)g * 8 + q];
                    val = (q == 3 || q == 4) ? rv : log2f(rv + 1.0f);
                }
                // f in [1115,1120): zero padding
            }
            frow[f] = f2bf(val);
        }
    }
    __syncthreads();

    // ---- layer 1: [64x1120] @ [1120x1024] + LN + ReLU -> sH ----
    mlp_layer(sFeat, FSTR, FPAD, Wt1, FPAD, b1, g1, be1, sH, HSTR, sSum, sSq, lane, wave);

    if (tid < 2 * ROWS) sSum[tid] = 0.0f;
    __syncthreads();

    // ---- layer 2: [64x1024] @ [1024x1024] + LN + ReLU -> sH2 (reuse sFeat LDS) ----
    bf16_t* sH2 = sFeat;
    mlp_layer(sH, HSTR, HN, Wt2, HN, b2, g2, be2, sH2, HSTR, sSum, sSq, lane, wave);

    // ---- final GEMV: out = h2 @ W3 + b3 ----
    {
        int row = tid >> 3, p = tid & 7;          // 8 threads per row
        const bf16_t* hrow = sH2 + (size_t)row * HSTR;
        float s = 0.0f;
        int k0 = p * 128;
        #pragma unroll 4
        for (int k = k0; k < k0 + 128; ++k) s += bf2f(hrow[k]) * W3[k];
        sRed[tid] = s;
        __syncthreads();
        if (p == 0 && (row0 + row) < nrows) {
            float tot = 0.0f;
            #pragma unroll
            for (int i = 0; i < 8; ++i) tot += sRed[(row << 3) + i];
            out[row0 + row] = tot + b3[0];
        }
    }
}

// ---------------- host launcher ----------------
extern "C" void kernel_launch(void* const* d_in, const int* in_sizes, int n_in,
                              void* d_out, int out_size, void* d_ws, size_t ws_size,
                              hipStream_t stream) {
    (void)n_in; (void)out_size; (void)ws_size;
    const int*   kc     = (const int*)  d_in[0];
    const int*   nl     = (const int*)  d_in[1];
    const float* op_vec = (const float*)d_in[2];
    const float* shints = (const float*)d_in[3];
    const float* rdf    = (const float*)d_in[4];
    const int*   rdb    = (const int*)  d_in[5];
    const float* wdf    = (const float*)d_in[6];
    const int*   wdb    = (const int*)  d_in[7];
    const float* rest   = (const float*)d_in[8];
    const float* Ekc    = (const float*)d_in[9];
    const float* Enl    = (const float*)d_in[10];
    const float* opW    = (const float*)d_in[11];
    const float* opB    = (const float*)d_in[12];
    const float* Ec     = (const float*)d_in[13];
    const float* Es     = (const float*)d_in[14];
    const float* Ei     = (const float*)d_in[15];
    const float* W1     = (const float*)d_in[16];
    const float* b1     = (const float*)d_in[17];
    const float* W2     = (const float*)d_in[18];
    const float* b2     = (const float*)d_in[19];
    const float* W3     = (const float*)d_in[20];
    const float* b3     = (const float*)d_in[21];
    const float* g1     = (const float*)d_in[22];
    const float* be1    = (const float*)d_in[23];
    const float* g2     = (const float*)d_in[24];
    const float* be2    = (const float*)d_in[25];

    const int nrows = in_sizes[0];                 // B

    bf16_t* Wt1 = (bf16_t*)d_ws;                                   // [1024][1120] bf16
    bf16_t* Wt2 = (bf16_t*)((char*)d_ws + (size_t)HN * FPAD * 2);  // [1024][1024] bf16

    convert_w1<<<(HN * FPAD + 255) / 256, 256, 0, stream>>>(W1, Wt1);
    convert_w2<<<(HN * HN + 255) / 256, 256, 0, stream>>>(W2, Wt2);

    const size_t shmem = (size_t)ROWS * FSTR * 2 + (size_t)ROWS * HSTR * 2
                       + 2 * ROWS * sizeof(float) + THREADS * sizeof(float);
    (void)hipFuncSetAttribute((const void*)fused_mlp,
                              hipFuncAttributeMaxDynamicSharedMemorySize, (int)shmem);

    fused_mlp<<<(nrows + ROWS - 1) / ROWS, THREADS, shmem, stream>>>(
        kc, nl, op_vec, shints, rdf, rdb, wdf, wdb, rest,
        Ekc, Enl, opW, opB, Ec, Es, Ei,
        b1, b2, W3, b3, g1, be1, g2, be2,
        Wt1, Wt2, (float*)d_out, nrows);
}
